// Projector_2216203125025
// MI455X (gfx1250) — matrix operand order repaired and checked by reading, not compiled, not run on hardware
//
#include <hip/hip_runtime.h>
#include <hip/hip_bf16.h>

// ---------------------------------------------------------------------------
// Fused MLP projector for MI455X (gfx1250, wave32, WMMA).
//   h   = emb @ W1 + b1            [16384, 4096]   (bf16 WMMA, f32 accum)
//   h   = GELU(LayerNorm(h))                        (f32 epilogue, exact erf)
//   out = h @ W2                   [16384, 1024]   (bf16 WMMA, f32 accum)
// Compute-bound (~275 GFLOP vs ~420 MB traffic) -> bf16 matrix path.
// Inner loops are software-pipelined: 4-deep rotating B-fragment register
// buffer + A-fragment double buffering, so WMMAs overlap L2/HBM latency.
// ---------------------------------------------------------------------------

#define B_ROWS   16384
#define D_IN     1024
#define D_H      4096
#define D_OUT    1024
#define LN_EPS   1e-5f

typedef __attribute__((ext_vector_type(16))) __bf16 v16bf;
typedef __attribute__((ext_vector_type(8)))  float  v8f;

// ---- fragment loaders (layouts per CDNA5 ISA 7.12.2, wave32) --------------
// A 16x32 bf16 tile: lane L -> row M = L&15; j0..7 = K (half*8 + 0..7),
// j8..15 = K (16 + half*8 + 0..7)  => two 16B chunks 32B apart.
__device__ __forceinline__ v16bf load_a_frag_bf16(const __bf16* base) {
    union { v16bf v; uint4 q[2]; } f;
    f.q[0] = *reinterpret_cast<const uint4*>(base);       // k +0..7
    f.q[1] = *reinterpret_cast<const uint4*>(base + 16);  // k +16..23
    return f.v;
}

__device__ __forceinline__ v16bf cvt_a_frag(float4 a0, float4 a1,
                                            float4 a2, float4 a3) {
    v16bf v;
    v[0]=(__bf16)a0.x;  v[1]=(__bf16)a0.y;  v[2]=(__bf16)a0.z;  v[3]=(__bf16)a0.w;
    v[4]=(__bf16)a1.x;  v[5]=(__bf16)a1.y;  v[6]=(__bf16)a1.z;  v[7]=(__bf16)a1.w;
    v[8]=(__bf16)a2.x;  v[9]=(__bf16)a2.y;  v[10]=(__bf16)a2.z; v[11]=(__bf16)a2.w;
    v[12]=(__bf16)a3.x; v[13]=(__bf16)a3.y; v[14]=(__bf16)a3.z; v[15]=(__bf16)a3.w;
    return v;
}

// B 32x16 bf16 tile: lane L -> N col = L&15; j0..7 = K (half*8 + 0..7),
// j8..15 = K (16 + half*8 + 0..7) => strided gather down row-major weight.
__device__ __forceinline__ v16bf load_b_frag(const __bf16* base, int ld) {
    v16bf v;
#pragma unroll
    for (int s = 0; s < 8; ++s) {
        v[s]     = base[(size_t)s * ld];
        v[s + 8] = base[(size_t)(16 + s) * ld];
    }
    return v;
}

// ---------------------------------------------------------------------------
// Kernel 0: f32 -> bf16 row-major conversion (8 elems/thread, b128 traffic)
// ---------------------------------------------------------------------------
__global__ void f32_to_bf16_vec8(const float* __restrict__ src,
                                 __bf16* __restrict__ dst, int n8) {
    int i = blockIdx.x * blockDim.x + threadIdx.x;
    if (i >= n8) return;
    const float4* s = reinterpret_cast<const float4*>(src) + 2 * (size_t)i;
    float4 a = s[0], b = s[1];
    union { __bf16 h[8]; uint4 q; } u;
    u.h[0]=(__bf16)a.x; u.h[1]=(__bf16)a.y; u.h[2]=(__bf16)a.z; u.h[3]=(__bf16)a.w;
    u.h[4]=(__bf16)b.x; u.h[5]=(__bf16)b.y; u.h[6]=(__bf16)b.z; u.h[7]=(__bf16)b.w;
    reinterpret_cast<uint4*>(dst)[i] = u.q;
}

// ---------------------------------------------------------------------------
// Kernel 1: GEMM1 + bias + LayerNorm + GELU(erf), H -> bf16.
// Grid: B/16 blocks of 512 threads (16 waves). One block owns a full
// 16-row x 4096-col stripe so LN stats stay on-chip (LDS ds_add_f32).
// Wave w covers columns [256w, 256w+256): 16 WMMA accum tiles (128 VGPRs).
// ---------------------------------------------------------------------------
__global__ __launch_bounds__(512) void gemm1_ln_gelu(
    const float* __restrict__ emb, const __bf16* __restrict__ w1,
    const float* __restrict__ b1, const float* __restrict__ gamma,
    const float* __restrict__ beta, __bf16* __restrict__ h) {
    __shared__ float s_sum[16], s_sq[16], s_mu[16], s_rs[16];

    const int lane = threadIdx.x & 31;
    const int wave = threadIdx.x >> 5;          // 0..15
    const int half = lane >> 4;                 // 0/1
    const int mrow = lane & 15;                 // A row / B col within tile
    const int krow = mrow + (half << 4);        // B K-row within 32-tile
    const int m0   = blockIdx.x << 4;
    const int n0   = wave << 8;                 // 256 cols per wave

    if (threadIdx.x < 16) { s_sum[threadIdx.x] = 0.f; s_sq[threadIdx.x] = 0.f; }
    __syncthreads();

    v8f acc[16];
    const v8f vzero = {0.f,0.f,0.f,0.f,0.f,0.f,0.f,0.f};
#pragma unroll
    for (int t = 0; t < 16; ++t) acc[t] = vzero;

    const float*  arow  = emb + (size_t)(m0 + mrow) * D_IN + half * 8;
    const __bf16* bbase = w1 + (size_t)(half * 8) * D_H + n0 + mrow;

    // ---- A double buffer: issue loads for k-step 0 ----
    float4 a0 = *reinterpret_cast<const float4*>(arow);
    float4 a1 = *reinterpret_cast<const float4*>(arow + 4);
    float4 a2 = *reinterpret_cast<const float4*>(arow + 16);
    float4 a3 = *reinterpret_cast<const float4*>(arow + 20);

    for (int ks = 0; ks < D_IN / 32; ++ks) {
        v16bf afrag = cvt_a_frag(a0, a1, a2, a3);
        if (ks + 1 < D_IN / 32) {            // issue next-k A loads early
            const float* np = arow + (ks + 1) * 32;
            a0 = *reinterpret_cast<const float4*>(np);
            a1 = *reinterpret_cast<const float4*>(np + 4);
            a2 = *reinterpret_cast<const float4*>(np + 16);
            a3 = *reinterpret_cast<const float4*>(np + 20);
            __builtin_prefetch(np + 32, 0, 0);   // global_prefetch_b8
        }
        const __bf16* bp = bbase + (size_t)ks * 32 * D_H;

        // ---- 4-deep rotating B-fragment pipeline (8 loads in flight) ----
        v16bf bf[4];
        bf[0] = load_b_frag(bp, D_H);
        bf[1] = load_b_frag(bp + 16, D_H);
        bf[2] = load_b_frag(bp + 32, D_H);
        bf[3] = load_b_frag(bp + 48, D_H);
#pragma unroll
        for (int t = 0; t < 16; ++t) {
            v16bf cur = bf[t & 3];
            if (t + 4 < 16) bf[t & 3] = load_b_frag(bp + (t + 4) * 16, D_H);
            acc[t] = __builtin_amdgcn_wmma_f32_16x16x32_bf16(
                false, afrag, false, cur, (short)0, acc[t], false, false);
        }
    }

    // ---- bias + per-row partial LN stats -> LDS float atomics ----
    float psum[8] = {0.f}, psq[8] = {0.f};
#pragma unroll
    for (int t = 0; t < 16; ++t) {
        float bias = b1[n0 + t * 16 + mrow];
#pragma unroll
        for (int v = 0; v < 8; ++v) {
            float x = acc[t][v] + bias;    // C layout: vgpr v -> row v+8*half
            acc[t][v] = x;
            psum[v] += x;
            psq[v]  += x * x;
        }
    }
#pragma unroll
    for (int v = 0; v < 8; ++v) {
        int r = v + (half << 3);
        atomicAdd(&s_sum[r], psum[v]);     // ds_add_f32
        atomicAdd(&s_sq[r],  psq[v]);
    }
    __syncthreads();
    if (threadIdx.x < 16) {
        float mu  = s_sum[threadIdx.x] * (1.f / D_H);
        float var = s_sq[threadIdx.x] * (1.f / D_H) - mu * mu;
        s_mu[threadIdx.x] = mu;
        s_rs[threadIdx.x] = rsqrtf(var + LN_EPS);
    }
    __syncthreads();

    // ---- LN + exact GELU, write H in bf16 row-major ----
#pragma unroll
    for (int t = 0; t < 16; ++t) {
        int n   = n0 + t * 16 + mrow;
        float g = gamma[n], bt = beta[n];
#pragma unroll
        for (int v = 0; v < 8; ++v) {
            int r   = v + (half << 3);
            float x = (acc[t][v] - s_mu[r]) * s_rs[r] * g + bt;
            x = 0.5f * x * (1.f + erff(x * 0.70710678118654752f));
            h[(size_t)(m0 + r) * D_H + n] = (__bf16)x;
        }
    }
}

// ---------------------------------------------------------------------------
// Kernel 2: out = H(bf16) @ W2(bf16), f32 accum/store.
// Grid: (B/32, D_OUT/512), 256 threads (8 waves as 2x4).
// Wave tile 16x128 -> 8 accum tiles. K = 4096 -> 128 k-steps.
// ---------------------------------------------------------------------------
__global__ __launch_bounds__(256) void gemm2(
    const __bf16* __restrict__ hbf, const __bf16* __restrict__ w2,
    float* __restrict__ out) {
    const int lane = threadIdx.x & 31;
    const int wave = threadIdx.x >> 5;          // 0..7
    const int wm   = wave >> 2, wn = wave & 3;
    const int half = lane >> 4;
    const int mrow = lane & 15;
    const int krow = mrow + (half << 4);
    const int m0   = blockIdx.x * 32 + wm * 16;
    const int n0   = blockIdx.y * 512 + wn * 128;

    v8f acc[8];
    const v8f vzero = {0.f,0.f,0.f,0.f,0.f,0.f,0.f,0.f};
#pragma unroll
    for (int t = 0; t < 8; ++t) acc[t] = vzero;

    const __bf16* arow  = hbf + (size_t)(m0 + mrow) * D_H + half * 8;
    const __bf16* bbase = w2 + (size_t)(half * 8) * D_OUT + n0 + mrow;

    // ---- A double buffer ----
    v16bf afrag = load_a_frag_bf16(arow);

    for (int ks = 0; ks < D_H / 32; ++ks) {
        v16bf acur = afrag;
        if (ks + 1 < D_H / 32) {             // issue next-k A loads early
            afrag = load_a_frag_bf16(arow + (ks + 1) * 32);
            __builtin_prefetch(arow + (ks + 1) * 32 + 32, 0, 0);
        }
        const __bf16* bp = bbase + (size_t)ks * 32 * D_OUT;

        // ---- 4-deep rotating B-fragment pipeline ----
        v16bf bf[4];
        bf[0] = load_b_frag(bp, D_OUT);
        bf[1] = load_b_frag(bp + 16, D_OUT);
        bf[2] = load_b_frag(bp + 32, D_OUT);
        bf[3] = load_b_frag(bp + 48, D_OUT);
#pragma unroll
        for (int t = 0; t < 8; ++t) {
            v16bf cur = bf[t & 3];
            if (t + 4 < 8) bf[t & 3] = load_b_frag(bp + (t + 4) * 16, D_OUT);
            acc[t] = __builtin_amdgcn_wmma_f32_16x16x32_bf16(
                false, acur, false, cur, (short)0, acc[t], false, false);
        }
    }

#pragma unroll
    for (int t = 0; t < 8; ++t) {
        int n = n0 + t * 16 + mrow;
#pragma unroll
        for (int v = 0; v < 8; ++v)
            out[(size_t)(m0 + v + (half << 3)) * D_OUT + n] = acc[t][v];
    }
}

// ---------------------------------------------------------------------------
extern "C" void kernel_launch(void* const* d_in, const int* in_sizes, int n_in,
                              void* d_out, int out_size, void* d_ws, size_t ws_size,
                              hipStream_t stream) {
    const float* emb   = (const float*)d_in[0];
    const float* W1    = (const float*)d_in[1];
    const float* b1    = (const float*)d_in[2];
    const float* gamma = (const float*)d_in[3];
    const float* beta  = (const float*)d_in[4];
    const float* W2    = (const float*)d_in[5];
    float* out = (float*)d_out;

    // workspace layout: W1bf (8 MiB) | W2bf (8 MiB) | Hbf (128 MiB)
    char* ws = (char*)d_ws;
    __bf16* w1bf = (__bf16*)ws;
    __bf16* w2bf = (__bf16*)(ws + (size_t)(D_IN * D_H) * 2);
    __bf16* hbf  = (__bf16*)(ws + (size_t)(D_IN * D_H) * 2 + (size_t)(D_H * D_OUT) * 2);

    const int n8_w1 = (D_IN * D_H) / 8;
    const int n8_w2 = (D_H * D_OUT) / 8;
    f32_to_bf16_vec8<<<(n8_w1 + 255) / 256, 256, 0, stream>>>(W1, w1bf, n8_w1);
    f32_to_bf16_vec8<<<(n8_w2 + 255) / 256, 256, 0, stream>>>(W2, w2bf, n8_w2);

    gemm1_ln_gelu<<<B_ROWS / 16, 512, 0, stream>>>(emb, w1bf, b1, gamma, beta, hbf);

    gemm2<<<dim3(B_ROWS / 32, D_OUT / 512), 256, 0, stream>>>(hbf, w2bf, out);
}
